// iA3Transformer_55319178772938
// MI455X (gfx1250) — matrix operand order repaired and checked by reading, not compiled, not run on hardware
//
#include <hip/hip_runtime.h>

// ---------- types ----------
typedef __bf16 bf16_t;
typedef __attribute__((ext_vector_type(16))) __bf16 v16bf;
typedef __attribute__((ext_vector_type(8)))  __bf16 bf16x8;
typedef __attribute__((ext_vector_type(8)))  float  v8f;

#define DD   768
#define SS   1024
#define BB   32
#define LL   12
#define NTOK (BB*SS)

union frag16 { v16bf v; bf16x8 h[2]; };

__device__ __forceinline__ v8f vzero8() {
  v8f z = {0.f,0.f,0.f,0.f,0.f,0.f,0.f,0.f};
  return z;
}

// A fragment (16x32 bf16): lanes 0-15 hold row M=lane, K={0..7,16..23};
// lanes 16-31 hold K={8..15,24..31}.
__device__ __forceinline__ v16bf load_a_frag(const bf16_t* row, int half) {
  frag16 f;
  f.h[0] = *(const bf16x8*)(row + half*8);
  f.h[1] = *(const bf16x8*)(row + 16 + half*8);
  return f.v;
}
// B fragment (32x16 bf16): per-lane = column N=lane&15; lanes 0-15 K={0..7,16..23},
// lanes 16-31 K={8..15,24..31} (same order as A, K-major source row).
__device__ __forceinline__ v16bf load_b_frag(const bf16_t* row, int half) {
  frag16 f;
  f.h[0] = *(const bf16x8*)(row + half*8);
  f.h[1] = *(const bf16x8*)(row + 16 + half*8);
  return f.v;
}
__device__ __forceinline__ v8f wmma_bf16(v16bf a, v16bf b, v8f c) {
  return __builtin_amdgcn_wmma_f32_16x16x32_bf16(false, a, false, b, (short)0, c,
                                                 false, false);
}

// CDNA5 async copy: 32 bytes (two B128) global -> LDS, tracked by ASYNCcnt.
// Same inst offset is added to BOTH the LDS and global addresses (ISA 10.7),
// matching our tiles (chunks +16B apart on both sides).
__device__ __forceinline__ void async_copy_row32(unsigned lds_off,
                                                 const bf16_t* gsrc) {
  unsigned long long ga = (unsigned long long)(uintptr_t)gsrc;
  asm volatile(
      "global_load_async_to_lds_b128 %0, %1, off\n\t"
      "global_load_async_to_lds_b128 %0, %1, off offset:16"
      :: "v"(lds_off), "v"(ga) : "memory");
}
__device__ __forceinline__ void wait_async4() {
  asm volatile("s_wait_asynccnt 0x4" ::: "memory");
}
__device__ __forceinline__ void wait_async0() {
  asm volatile("s_wait_asynccnt 0x0" ::: "memory");
}

// ---------- prep: f32 -> bf16 hidden ----------
__global__ __launch_bounds__(256) void cvt_kernel(const float* __restrict__ src,
                                                  bf16_t* __restrict__ dst, int n) {
  int base = (blockIdx.x * 256 + threadIdx.x) * 4;
  if (base + 3 < n) {
#pragma unroll
    for (int i = 0; i < 4; ++i) dst[base + i] = (bf16_t)src[base + i];
  }
}

// ---------- prep: transpose+cast weights; Wt[l][g][n][k] = W_g[l][k][n] ----------
__global__ __launch_bounds__(256) void wprep_kernel(const float* __restrict__ Wq,
                                                    const float* __restrict__ Wk,
                                                    const float* __restrict__ Wv,
                                                    bf16_t* __restrict__ Wt) {
  size_t i = (size_t)blockIdx.x * 256 + threadIdx.x;
  if (i >= (size_t)LL * 3 * DD * DD) return;
  int kd = (int)(i % DD);
  size_t r = i / DD;
  int n = (int)(r % DD); r /= DD;
  int g = (int)(r % 3);
  int l = (int)(r / 3);
  const float* W = (g == 0) ? Wq : (g == 1) ? Wk : Wv;
  Wt[i] = (bf16_t)W[((size_t)l * DD + kd) * DD + n];
}

// ---------- generic WMMA GEMM ----------
// WG = 256 threads (8 waves), tile 128x128, wave tile 32x64 (2x4 WMMA frags).
// A: [M x K] row-major (lda), B: [N x K] row-major (ldb, pre-transposed).
// Double-buffered LDS staged with GLOBAL_LOAD_ASYNC_TO_LDS_B128, peeled tail.
// MODE 0: Q  -> bf16 out (ld 768), (acc+bias[N]) * 1/sqrt(768)  [softmax scale]
// MODE 1: K  -> bf16 out (ld 768), (acc+bias[N])*cscale[N]
// MODE 2: V  -> bf16 out TRANSPOSED [b][n][s], (acc+bias[N])*cscale[N]
// MODE 3: S  -> f32 out (ld 1024), raw acc (scale folded into Q)
// MODE 4: PV -> bf16 out (ld 768), acc * (1/rowsum[M])
template <int MODE>
__global__ __launch_bounds__(256) void gemm_kernel(
    const bf16_t* __restrict__ A, size_t lda, size_t sAz,
    const bf16_t* __restrict__ B, size_t ldb, size_t sBz,
    int Kdim,
    void* __restrict__ Out, size_t sOz,
    const float* __restrict__ bias,
    const float* __restrict__ cscale,
    const float* __restrict__ rowsum, size_t sRz)
{
  __shared__ __align__(16) bf16_t lA[2][128 * 32];
  __shared__ __align__(16) bf16_t lB[2][128 * 32];
  const int m0 = blockIdx.x * 128;
  const int n0 = blockIdx.y * 128;
  const int z  = blockIdx.z;
  const int t    = threadIdx.x;
  const int wave = t >> 5;
  const int lane = t & 31;
  const int half = lane >> 4;
  const int lrow = lane & 15;
  const int wm   = wave & 3;   // 4 row-strips of 32
  const int wn   = wave >> 2;  // 2 col-strips of 64
  const int r = t >> 1;        // staging row 0..127
  const int c = (t & 1) * 16;  // staging col 0 / 16

  const bf16_t* Ag = A + sAz * z + (size_t)(m0 + r) * lda + c;
  const bf16_t* Bg = B + sBz * z + (size_t)(n0 + r) * ldb + c;
  const unsigned ldsA0 = (unsigned)(uintptr_t)&lA[0][r * 32 + c];
  const unsigned ldsB0 = (unsigned)(uintptr_t)&lB[0][r * 32 + c];
  const unsigned bufB  = 128 * 32 * 2;  // bytes per buffer

  v8f acc[2][4];
#pragma unroll
  for (int mi = 0; mi < 2; ++mi)
#pragma unroll
    for (int ni = 0; ni < 4; ++ni) acc[mi][ni] = vzero8();

  const bf16_t* aFragBase = &lA[0][(wm * 32 + lrow) * 32];
  const bf16_t* bFragBase = &lB[0][(wn * 64 + lrow) * 32];
  const unsigned bufE = 128 * 32;  // elements per buffer

  // prologue: stage k=0 into buffer 0
  async_copy_row32(ldsA0, Ag);
  async_copy_row32(ldsB0, Bg);
  Ag += 32; Bg += 32;

  int buf = 0;
  // steady state: branch-free stage -> wait -> barrier -> 8 WMMA -> barrier
  for (int k0 = 32; k0 < Kdim; k0 += 32, buf ^= 1) {
    async_copy_row32(ldsA0 + (buf ^ 1) * bufB, Ag);
    async_copy_row32(ldsB0 + (buf ^ 1) * bufB, Bg);
    Ag += 32; Bg += 32;
    wait_async4();   // current buffer's 4 loads complete (in-order)
    __syncthreads(); // all waves staged their part of current buffer
    {
      const bf16_t* aB = aFragBase + buf * bufE;
      const bf16_t* bB = bFragBase + buf * bufE;
      v16bf a0 = load_a_frag(aB, half);
      v16bf a1 = load_a_frag(aB + 16 * 32, half);
#pragma unroll
      for (int ni = 0; ni < 4; ++ni) {
        v16bf bb = load_b_frag(bB + ni * 16 * 32, half);
        acc[0][ni] = wmma_bf16(a0, bb, acc[0][ni]);
        acc[1][ni] = wmma_bf16(a1, bb, acc[1][ni]);
      }
    }
    __syncthreads(); // all waves done reading current buffer
  }
  // peeled last k-step
  wait_async0();
  __syncthreads();
  {
    const bf16_t* aB = aFragBase + buf * bufE;
    const bf16_t* bB = bFragBase + buf * bufE;
    v16bf a0 = load_a_frag(aB, half);
    v16bf a1 = load_a_frag(aB + 16 * 32, half);
#pragma unroll
    for (int ni = 0; ni < 4; ++ni) {
      v16bf bb = load_b_frag(bB + ni * 16 * 32, half);
      acc[0][ni] = wmma_bf16(a0, bb, acc[0][ni]);
      acc[1][ni] = wmma_bf16(a1, bb, acc[1][ni]);
    }
  }

  // ---- epilogue ----
  // C layout: VGPR i -> M = base + i + 8*half, N = lane&15
  if constexpr (MODE == 3) {
    float* O = (float*)Out + sOz * z;
#pragma unroll
    for (int mi = 0; mi < 2; ++mi) {
      const int Mb = m0 + wm * 32 + mi * 16 + 8 * half;
#pragma unroll
      for (int ni = 0; ni < 4; ++ni) {
        const int N = n0 + wn * 64 + ni * 16 + lrow;
#pragma unroll
        for (int i = 0; i < 8; ++i)
          O[(size_t)(Mb + i) * SS + N] = acc[mi][ni][i];
      }
    }
  } else if constexpr (MODE == 4) {
    bf16_t* O = (bf16_t*)Out + sOz * z;
    const float* rsb = rowsum + sRz * z;
#pragma unroll
    for (int mi = 0; mi < 2; ++mi) {
      const int Mb = m0 + wm * 32 + mi * 16 + 8 * half;
      float rinv[8];
#pragma unroll
      for (int i = 0; i < 8; ++i) rinv[i] = 1.f / rsb[Mb + i];
#pragma unroll
      for (int ni = 0; ni < 4; ++ni) {
        const int N = n0 + wn * 64 + ni * 16 + lrow;
#pragma unroll
        for (int i = 0; i < 8; ++i)
          O[(size_t)(Mb + i) * DD + N] = (bf16_t)(acc[mi][ni][i] * rinv[i]);
      }
    }
  } else if constexpr (MODE == 2) {
    bf16_t* O = (bf16_t*)Out;
#pragma unroll
    for (int mi = 0; mi < 2; ++mi) {
      const int Mb = m0 + wm * 32 + mi * 16 + 8 * half;  // 8 consecutive tokens
      const int batch = Mb >> 10;
      const int srow  = Mb & 1023;
#pragma unroll
      for (int ni = 0; ni < 4; ++ni) {
        const int N = n0 + wn * 64 + ni * 16 + lrow;
        const float bv = bias[N], sv = cscale[N];
        bf16x8 vv;
#pragma unroll
        for (int i = 0; i < 8; ++i) vv[i] = (bf16_t)((acc[mi][ni][i] + bv) * sv);
        *(bf16x8*)&O[((size_t)batch * DD + N) * SS + srow] = vv;
      }
    }
  } else {  // MODE 0 / 1
    bf16_t* O = (bf16_t*)Out;
    const float qscale = 0.03608439182435161f;  // 1/sqrt(768), folded into Q
#pragma unroll
    for (int mi = 0; mi < 2; ++mi) {
      const int Mb = m0 + wm * 32 + mi * 16 + 8 * half;
#pragma unroll
      for (int ni = 0; ni < 4; ++ni) {
        const int N = n0 + wn * 64 + ni * 16 + lrow;
        const float bv = bias[N];
        const float sv = (MODE == 1) ? cscale[N] : qscale;
#pragma unroll
        for (int i = 0; i < 8; ++i)
          O[(size_t)(Mb + i) * DD + N] = (bf16_t)((acc[mi][ni][i] + bv) * sv);
      }
    }
  }
}

// ---------- row softmax: one wave per 1024-wide row ----------
// Writes exp(s - rowmax) bf16 IN PLACE over the f32 row (first half) and the
// row sum; 1/rowsum is folded into the PV epilogue.
__global__ __launch_bounds__(256) void softmax_kernel(float* __restrict__ S,
                                                      float* __restrict__ rowsum)
{
  const int wave = threadIdx.x >> 5, lane = threadIdx.x & 31;
  const int row = blockIdx.x * 8 + wave;  // 0..32767
  float* srow = S + (size_t)row * SS;
  float v[32];
#pragma unroll
  for (int j = 0; j < 32; ++j) v[j] = srow[j * 32 + lane];
  float m = -3.4e38f;
#pragma unroll
  for (int j = 0; j < 32; ++j) m = fmaxf(m, v[j]);
#pragma unroll
  for (int o = 16; o > 0; o >>= 1) m = fmaxf(m, __shfl_xor(m, o, 32));
  float s = 0.f;
#pragma unroll
  for (int j = 0; j < 32; ++j) { v[j] = __expf(v[j] - m); s += v[j]; }
#pragma unroll
  for (int o = 16; o > 0; o >>= 1) s += __shfl_xor(s, o, 32);
  bf16_t* prow = (bf16_t*)S + (size_t)row * 2048;  // bf16 row, stride 2048
#pragma unroll
  for (int j = 0; j < 32; ++j) prow[j * 32 + lane] = (bf16_t)v[j];
  if (lane == 0) rowsum[row] = s;
}

// ---------- head: out[b] = h[b,0,:] . W_head + b_head ----------
__global__ __launch_bounds__(64) void head_kernel(const bf16_t* __restrict__ H,
                                                  const float* __restrict__ Wh,
                                                  const float* __restrict__ bh,
                                                  float* __restrict__ out) {
  const int b = blockIdx.x;
  const int t = threadIdx.x;  // 64 threads
  const bf16_t* hrow = H + (size_t)b * SS * DD;  // token 0
  float s = 0.f;
  for (int d = t; d < DD; d += 64) s += (float)hrow[d] * Wh[d];
  __shared__ float red[64];
  red[t] = s;
  __syncthreads();
  for (int o = 32; o > 0; o >>= 1) {
    if (t < o) red[t] += red[t + o];
    __syncthreads();
  }
  if (t == 0) out[b] = red[0] + bh[0];
}

// ---------- launch ----------
extern "C" void kernel_launch(void* const* d_in, const int* in_sizes, int n_in,
                              void* d_out, int out_size, void* d_ws, size_t ws_size,
                              hipStream_t stream) {
  const float* hs = (const float*)d_in[0];
  const float* Wq = (const float*)d_in[1];
  const float* bq = (const float*)d_in[2];
  const float* Wk = (const float*)d_in[3];
  const float* bk = (const float*)d_in[4];
  const float* Wv = (const float*)d_in[5];
  const float* bv = (const float*)d_in[6];
  const float* lk = (const float*)d_in[7];
  const float* lv = (const float*)d_in[8];
  const float* Wh = (const float*)d_in[9];
  const float* bh = (const float*)d_in[10];

  char* ws = (char*)d_ws;
  size_t off = 0;
  bf16_t* hb = (bf16_t*)(ws + off); off += (size_t)NTOK * DD * 2;
  bf16_t* Qb = (bf16_t*)(ws + off); off += (size_t)NTOK * DD * 2;
  bf16_t* Kb = (bf16_t*)(ws + off); off += (size_t)NTOK * DD * 2;
  bf16_t* Vt = (bf16_t*)(ws + off); off += (size_t)NTOK * DD * 2;
  bf16_t* Wt = (bf16_t*)(ws + off); off += (size_t)LL * 3 * DD * DD * 2;
  float*  Sb = (float*)(ws + off);  off += (size_t)BB * SS * SS * 4;
  float*  rs = (float*)(ws + off);  off += (size_t)NTOK * 4;

  cvt_kernel<<<(NTOK * DD) / 1024, 256, 0, stream>>>(hs, hb, NTOK * DD);
  wprep_kernel<<<((size_t)LL * 3 * DD * DD + 255) / 256, 256, 0, stream>>>(
      Wq, Wk, Wv, Wt);

  const dim3 gQKV(NTOK / 128, DD / 128, 1);
  const dim3 gS(SS / 128, SS / 128, BB);
  const dim3 gPV(SS / 128, DD / 128, BB);

  for (int l = 0; l < LL; ++l) {
    const bf16_t* W = Wt + (size_t)l * 3 * DD * DD;
    gemm_kernel<0><<<gQKV, 256, 0, stream>>>(
        hb, DD, 0, W + 0 * (size_t)DD * DD, DD, 0, DD,
        Qb, 0, bq + l * DD, nullptr, nullptr, 0);
    gemm_kernel<1><<<gQKV, 256, 0, stream>>>(
        hb, DD, 0, W + 1 * (size_t)DD * DD, DD, 0, DD,
        Kb, 0, bk + l * DD, lk + l * DD, nullptr, 0);
    gemm_kernel<2><<<gQKV, 256, 0, stream>>>(
        hb, DD, 0, W + 2 * (size_t)DD * DD, DD, 0, DD,
        Vt, 0, bv + l * DD, lv + l * DD, nullptr, 0);
    gemm_kernel<3><<<gS, 256, 0, stream>>>(
        Qb, DD, (size_t)SS * DD, Kb, DD, (size_t)SS * DD, DD,
        Sb, (size_t)SS * SS, nullptr, nullptr, nullptr, 0);
    softmax_kernel<<<NTOK / 8, 256, 0, stream>>>(Sb, rs);
    gemm_kernel<4><<<gPV, 256, 0, stream>>>(
        (const bf16_t*)Sb, 2048, (size_t)SS * 2048, Vt, SS, (size_t)DD * SS, SS,
        hb, (size_t)SS * DD, nullptr, nullptr, rs, SS);
  }
  head_kernel<<<BB, 64, 0, stream>>>(hb, Wh, bh, (float*)d_out);
}
